// TransformerBlock_33036888441544
// MI455X (gfx1250) — matrix operand order repaired and checked by reading, not compiled, not run on hardware
//
#include <hip/hip_runtime.h>
#include <hip/hip_bf16.h>
#include <stdint.h>

#define B_   2
#define T_   4096
#define D_   512
#define H_   8
#define Kh   64
#define F_   2048
#define BT   (B_ * T_)        // 8192 rows
#define QKVN (3 * D_)         // 1536

typedef __bf16 bf16;
typedef __bf16 v16bf __attribute__((ext_vector_type(16)));
typedef __bf16 v8bf  __attribute__((ext_vector_type(8)));
typedef float  v8f   __attribute__((ext_vector_type(8)));
typedef unsigned int u32x4 __attribute__((ext_vector_type(4)));
typedef int          i32x4 __attribute__((ext_vector_type(4)));
typedef int          i32x8 __attribute__((ext_vector_type(8)));

// ------------------------------------------------------- fast transcendentals
__device__ static inline float fast_exp2(float x) {
#if __has_builtin(__builtin_amdgcn_exp2f)
  return __builtin_amdgcn_exp2f(x);   // native v_exp_f32 (base-2), no fixup
#else
  return exp2f(x);
#endif
}
__device__ static inline float fast_rcp(float x) {
#if __has_builtin(__builtin_amdgcn_rcpf)
  return __builtin_amdgcn_rcpf(x);
#else
  return 1.0f / x;
#endif
}
__device__ static inline float fast_tanh(float x) {
#if __has_builtin(__builtin_amdgcn_tanhf)
  return __builtin_amdgcn_tanhf(x);   // native v_tanh_f32 on gfx1250
#elif __has_builtin(__builtin_amdgcn_tanh_f32)
  return __builtin_amdgcn_tanh_f32(x);
#else
  // branch-free: tanh(x) = 1 - 2/(1 + exp2(2*log2(e)*x))
  float e = fast_exp2(x * 2.88539008178f);
  return 1.0f - 2.0f * fast_rcp(e + 1.0f);
#endif
}

// ---------------------------------------------------------------- WMMA helpers
__device__ static inline v8f wmma_bf16(v16bf a, v16bf b, v8f c) {
  return __builtin_amdgcn_wmma_f32_16x16x32_bf16(false, a, false, b, (short)0, c,
                                                 false, false);
}

// Build a v16bf A/B fragment from two 16-byte runs (or one 32-byte run).
__device__ static inline v16bf ld16_2(const bf16* p0, const bf16* p1) {
  v8bf a = *(const v8bf*)p0;
  v8bf b = *(const v8bf*)p1;
  v16bf r;
#pragma unroll
  for (int i = 0; i < 8; i++) { r[i] = a[i]; r[i + 8] = b[i]; }
  return r;
}
__device__ static inline v16bf ld16_1(const bf16* p) { return ld16_2(p, p + 8); }

__device__ static inline uint32_t lds_off(const void* p) {
  // generic (flat) LDS pointer: low 32 bits are the wave-relative LDS offset
  return (uint32_t)(uintptr_t)p;
}

// ------------------------------------------------- Tensor Data Mover (2D tile)
// data_size = 2 bytes (bf16). Optional LDS padding: 4 DWORDs after every 16
// DWORDs (64B data row -> 80B LDS row) to keep 16B alignment + spread banks.
__device__ static inline void tdm_load_2d(uint32_t ldsaddr, const void* gptr,
                                          uint32_t tile_w, uint32_t tile_h,
                                          uint32_t tensor_w, uint32_t tensor_h,
                                          uint64_t row_stride_elems,
                                          int padded) {
  uint64_t ga = (uint64_t)(uintptr_t)gptr;
  u32x4 g0;
  g0[0] = 1u;                                    // count=1, user mode
  g0[1] = ldsaddr;                               // lds_addr
  g0[2] = (uint32_t)ga;                          // global_addr[31:0]
  g0[3] = (uint32_t)((ga >> 32) & 0x1FFFFFFu) |  // global_addr[56:32]
          (2u << 30);                            // type = 2 (image)
  i32x8 g1;
  uint32_t w0 = (1u << 16);                      // data_size = 1 -> 2 bytes
  if (padded) w0 |= (1u << 20) | (3u << 22) | (3u << 25); // pad 16DW -> +4DW
  g1[0] = (int)w0;
  g1[1] = (int)((tensor_w & 0xFFFFu) << 16);                       // dim0 lo
  g1[2] = (int)((tensor_w >> 16) | ((tensor_h & 0xFFFFu) << 16));  // dim0 hi|dim1 lo
  g1[3] = (int)((tensor_h >> 16) | ((tile_w & 0xFFFFu) << 16));    // dim1 hi|tile0
  g1[4] = (int)(tile_h & 0xFFFFu);                                 // tile1, tile2=0
  g1[5] = (int)(uint32_t)(row_stride_elems & 0xFFFFFFFFull);       // stride0 lo
  g1[6] = (int)(uint32_t)((row_stride_elems >> 32) & 0xFFFFull);   // stride0 hi
  g1[7] = 0;
  i32x4 z4 = {0, 0, 0, 0};
#if __clang_major__ >= 23
  i32x8 z8 = {0, 0, 0, 0, 0, 0, 0, 0};
  __builtin_amdgcn_tensor_load_to_lds(g0, g1, z4, z4, z8, 0);
#else
  __builtin_amdgcn_tensor_load_to_lds(g0, g1, z4, z4, 0);
#endif
}

// -------------------------------------------------------------- small kernels
// fp32 [K][N] -> bf16 [N][K] (transpose + convert)
__global__ void tconv_kernel(const float* __restrict__ src, bf16* __restrict__ dst,
                             int K, int N) {
  size_t i = (size_t)blockIdx.x * 256 + threadIdx.x;
  if (i >= (size_t)K * N) return;
  int n = (int)(i % N), k = (int)(i / N);
  dst[(size_t)n * K + k] = (bf16)src[i];
}

__global__ void bcat_kernel(const float* bq, const float* bk, const float* bv,
                            float* out) {
  int i = blockIdx.x * 256 + threadIdx.x;
  if (i < 1536)
    out[i] = (i < 512) ? bq[i] : (i < 1024 ? bk[i - 512] : bv[i - 1024]);
}

// qkv[(b*T+t)*1536 + 1024 + h*64 + d]  ->  vT[((b*8+h)*64+d)*4096 + t]
__global__ void vtrans_kernel(const bf16* __restrict__ qkv, bf16* __restrict__ vT) {
  int i = blockIdx.x * 256 + threadIdx.x;   // over 16*64*4096
  int t  = i & (T_ - 1);
  int d  = (i >> 12) & 63;
  int bh = i >> 18;
  int b = bh >> 3, h = bh & 7;
  vT[i] = qkv[((size_t)(b * T_ + t)) * QKVN + D_ * 2 + h * Kh + d];
}

// LayerNorm: fp32 [rows][512] -> bf16
__global__ __launch_bounds__(256) void ln_kernel(const float* __restrict__ x,
                                                 const float* __restrict__ sc,
                                                 const float* __restrict__ of,
                                                 bf16* __restrict__ out) {
  __shared__ float sh[256];
  const int row = blockIdx.x, tid = threadIdx.x;
  const float* xr = x + (size_t)row * D_;
  float a = xr[tid], b = xr[tid + 256];
  sh[tid] = a + b;
  __syncthreads();
  for (int st = 128; st > 0; st >>= 1) {
    if (tid < st) sh[tid] += sh[tid + st];
    __syncthreads();
  }
  float mean = sh[0] * (1.0f / D_);
  __syncthreads();
  sh[tid] = a * a + b * b;
  __syncthreads();
  for (int st = 128; st > 0; st >>= 1) {
    if (tid < st) sh[tid] += sh[tid + st];
    __syncthreads();
  }
  float var = sh[0] * (1.0f / D_) - mean * mean;
  float inv = rsqrtf(var + 1e-5f);
  out[(size_t)row * D_ + tid]       = (bf16)((a - mean) * inv * sc[tid] + of[tid]);
  out[(size_t)row * D_ + tid + 256] = (bf16)((b - mean) * inv * sc[tid + 256] + of[tid + 256]);
}

// ------------------------------------------------------------------- GEMM
// C[M,N] = act(A[M,K] @ Bt[N,K]^T + bias) (+ residual). A,Bt bf16; acc fp32.
// 128x128 workgroup tile, 8 waves (4x2), per-wave 32x64 = 2x4 WMMA tiles.
// TDM double-buffered LDS staging, 80-byte padded rows.
#define RS 80
__global__ __launch_bounds__(256) void gemm_bf16_kernel(
    const bf16* __restrict__ A, const bf16* __restrict__ Bt,
    const float* __restrict__ bias, const float* __restrict__ residual,
    float* __restrict__ outF, bf16* __restrict__ outH,
    int M, int N, int Kd, int act) {
  __shared__ __align__(16) char smA[2][128 * RS];
  __shared__ __align__(16) char smB[2][128 * RS];
  const int wave = threadIdx.x >> 5, lane = threadIdx.x & 31;
  const int l16 = lane & 15, half = lane >> 4;
  const int wm = wave >> 1, wn = wave & 1;
  const int mblk = blockIdx.y * 128, nblk = blockIdx.x * 128;
  const int nk = Kd / 32;

  v8f acc[2][4] = {};

  auto issue = [&](int bufI, int kt) {
    tdm_load_2d(lds_off(&smA[bufI][0]), A + (size_t)mblk * Kd + (size_t)kt * 32,
                32, 128, (uint32_t)Kd, (uint32_t)M, (uint64_t)Kd, 1);
    tdm_load_2d(lds_off(&smB[bufI][0]), Bt + (size_t)nblk * Kd + (size_t)kt * 32,
                32, 128, (uint32_t)Kd, (uint32_t)N, (uint64_t)Kd, 1);
  };

  if (threadIdx.x < 32) {
    issue(0, 0);
    __builtin_amdgcn_s_wait_tensorcnt(0);
  }
  __syncthreads();

  for (int kt = 0; kt < nk; ++kt) {
    const int buf = kt & 1;
    if (threadIdx.x < 32 && kt + 1 < nk) issue(buf ^ 1, kt + 1);

    const char* Ab = &smA[buf][0];
    const char* Bb = &smB[buf][0];
    v16bf af[2], bfr[4];
#pragma unroll
    for (int i = 0; i < 2; i++) {
      const char* p = Ab + (size_t)(wm * 32 + i * 16 + l16) * RS + 32 * half;
      af[i] = ld16_1((const bf16*)p);
    }
#pragma unroll
    for (int j = 0; j < 4; j++) {
      const char* p = Bb + (size_t)(wn * 64 + j * 16 + l16) * RS + 32 * half;
      bfr[j] = ld16_1((const bf16*)p);
    }
#pragma unroll
    for (int i = 0; i < 2; i++)
#pragma unroll
      for (int j = 0; j < 4; j++) acc[i][j] = wmma_bf16(af[i], bfr[j], acc[i][j]);

    if (threadIdx.x < 32 && kt + 1 < nk) __builtin_amdgcn_s_wait_tensorcnt(0);
    __syncthreads();
  }

  // epilogue: C row = r + 8*half, col = lane&15 within each 16x16 tile
#pragma unroll
  for (int i = 0; i < 2; i++)
#pragma unroll
    for (int j = 0; j < 4; j++) {
      const int col = nblk + wn * 64 + j * 16 + l16;
      const float bc = bias[col];
#pragma unroll
      for (int r = 0; r < 8; r++) {
        const int row = mblk + wm * 32 + i * 16 + 8 * half + r;
        float v = acc[i][j][r] + bc;
        if (act == 1) {  // tanh-approx GELU (jax.nn.gelu default)
          v = 0.5f * v *
              (1.0f + fast_tanh(0.7978845608f * (v + 0.044715f * v * v * v)));
        }
        const size_t idx = (size_t)row * N + col;
        if (residual) v += residual[idx];
        if (outF) outF[idx] = v;
        if (outH) outH[idx] = (bf16)v;
      }
    }
}

// --------------------------------------------------------------- attention
// Flash attention, per wave: 16 Q rows, keys in chunks of 32.
// Softmax runs in the exp2 domain: logits pre-scaled by 0.125*log2(e), so every
// exponential is a single native v_exp_f32 (exp2) with no range-fixup code.
#define SM_SCALE 0.18033688011f   // (1/sqrt(64)) * log2(e)
__global__ __launch_bounds__(256) void attn_kernel(const bf16* __restrict__ qkv,
                                                   const bf16* __restrict__ vT,
                                                   bf16* __restrict__ ctx) {
  __shared__ __align__(16) bf16 pbuf[8][16 * 40];  // per-wave P staging (padded)
  const int wave = threadIdx.x >> 5, lane = threadIdx.x & 31;
  const int l16 = lane & 15, half = lane >> 4;
  const int bh = blockIdx.x >> 5;       // 32 q-blocks per (b,h)
  const int qblk = blockIdx.x & 31;
  const int b = bh >> 3, h = bh & 7;
  const int qrow = qblk * 128 + wave * 16;

  const bf16* qp = qkv + ((size_t)(b * T_ + qrow)) * QKVN + h * Kh;
  const bf16* kp = qkv + (size_t)b * T_ * QKVN + D_ + h * Kh;
  const bf16* vp = vT + (size_t)bh * Kh * T_;
  bf16* pb = &pbuf[wave][0];

  // Q A-fragments (head dim 64 = 2 chunks of 32)
  v16bf aq[2];
  {
    const bf16* pr = qp + (size_t)l16 * QKVN;
    aq[0] = ld16_1(pr + 16 * half);
    aq[1] = ld16_1(pr + 32 + 16 * half);
  }

  float mrow[8], lrow[8];
  v8f o0 = {}, o1 = {}, o2 = {}, o3 = {};
#pragma unroll
  for (int r = 0; r < 8; r++) { mrow[r] = -3.0e38f; lrow[r] = 0.0f; }

  for (int kt = 0; kt < T_ / 32; ++kt) {
    const int key0 = kt * 32;
    // prefetch next chunk's K rows / V columns (global_prefetch_b8)
    if (kt + 1 < T_ / 32) {
      __builtin_prefetch(kp + (size_t)(key0 + 32 + l16) * QKVN + 16 * half, 0, 1);
      __builtin_prefetch(vp + (size_t)(l16)*T_ + key0 + 32 + 16 * half, 0, 1);
    }
    v8f s0 = {}, s1 = {};
#pragma unroll
    for (int c = 0; c < 2; c++) {
      v16bf bk0 = ld16_1(kp + (size_t)(key0 + l16) * QKVN + c * 32 + 16 * half);
      v16bf bk1 = ld16_1(kp + (size_t)(key0 + 16 + l16) * QKVN + c * 32 + 16 * half);
      s0 = wmma_bf16(aq[c], bk0, s0);
      s1 = wmma_bf16(aq[c], bk1, s1);
    }
    // online softmax over 32 keys; row = r + 8*half, col = lane&15
    float alpha[8];
#pragma unroll
    for (int r = 0; r < 8; r++) {
      float v0 = s0[r] * SM_SCALE, v1 = s1[r] * SM_SCALE;
      float mx = fmaxf(v0, v1);
      mx = fmaxf(mx, __shfl_xor(mx, 1, 32));
      mx = fmaxf(mx, __shfl_xor(mx, 2, 32));
      mx = fmaxf(mx, __shfl_xor(mx, 4, 32));
      mx = fmaxf(mx, __shfl_xor(mx, 8, 32));
      float mn = fmaxf(mrow[r], mx);
      alpha[r] = fast_exp2(mrow[r] - mn);
      mrow[r] = mn;
      float p0 = fast_exp2(v0 - mn), p1 = fast_exp2(v1 - mn);
      float sum = p0 + p1;
      sum += __shfl_xor(sum, 1, 32);
      sum += __shfl_xor(sum, 2, 32);
      sum += __shfl_xor(sum, 4, 32);
      sum += __shfl_xor(sum, 8, 32);
      lrow[r] = lrow[r] * alpha[r] + sum;
      pb[(r + 8 * half) * 40 + l16]      = (bf16)p0;
      pb[(r + 8 * half) * 40 + 16 + l16] = (bf16)p1;
    }
    asm volatile("s_wait_dscnt 0" ::: "memory");  // P store -> A-frag reload
    v16bf ap = ld16_1(pb + (size_t)l16 * 40 + 16 * half);
#pragma unroll
    for (int r = 0; r < 8; r++) {
      o0[r] *= alpha[r]; o1[r] *= alpha[r]; o2[r] *= alpha[r]; o3[r] *= alpha[r];
    }
    o0 = wmma_bf16(ap, ld16_1(vp + (size_t)(0 * 16 + l16) * T_ + key0 + 16 * half), o0);
    o1 = wmma_bf16(ap, ld16_1(vp + (size_t)(1 * 16 + l16) * T_ + key0 + 16 * half), o1);
    o2 = wmma_bf16(ap, ld16_1(vp + (size_t)(2 * 16 + l16) * T_ + key0 + 16 * half), o2);
    o3 = wmma_bf16(ap, ld16_1(vp + (size_t)(3 * 16 + l16) * T_ + key0 + 16 * half), o3);
  }

  const size_t obase = ((size_t)(b * T_ + qrow)) * D_ + h * Kh;
#pragma unroll
  for (int r = 0; r < 8; r++) {
    const float invl = fast_rcp(lrow[r]);
    const size_t rb = obase + (size_t)(r + 8 * half) * D_;
    ctx[rb + 0  + l16] = (bf16)(o0[r] * invl);
    ctx[rb + 16 + l16] = (bf16)(o1[r] * invl);
    ctx[rb + 32 + l16] = (bf16)(o2[r] * invl);
    ctx[rb + 48 + l16] = (bf16)(o3[r] * invl);
  }
}

// ----------------------------------------------------------------- launcher
extern "C" void kernel_launch(void* const* d_in, const int* in_sizes, int n_in,
                              void* d_out, int out_size, void* d_ws, size_t ws_size,
                              hipStream_t stream) {
  (void)in_sizes; (void)n_in; (void)out_size; (void)ws_size;
  const float* inputs = (const float*)d_in[0];
  const float* ln1_s  = (const float*)d_in[1];
  const float* ln1_o  = (const float*)d_in[2];
  const float* wq = (const float*)d_in[3];
  const float* bq = (const float*)d_in[4];
  const float* wk = (const float*)d_in[5];
  const float* bk = (const float*)d_in[6];
  const float* wv = (const float*)d_in[7];
  const float* bv = (const float*)d_in[8];
  const float* wo = (const float*)d_in[9];
  const float* bo = (const float*)d_in[10];
  const float* ln2_s = (const float*)d_in[11];
  const float* ln2_o = (const float*)d_in[12];
  const float* w1 = (const float*)d_in[13];
  const float* b1 = (const float*)d_in[14];
  const float* w2 = (const float*)d_in[15];
  const float* b2 = (const float*)d_in[16];
  float* out = (float*)d_out;

  char* ws = (char*)d_ws;
  size_t off = 0;
  auto alloc = [&](size_t bytes) -> char* {
    char* p = ws + off;
    off = (off + bytes + 255) & ~(size_t)255;
    return p;
  };
  bf16* wqkv_t = (bf16*)alloc((size_t)QKVN * D_ * 2);
  bf16* wo_t   = (bf16*)alloc((size_t)D_ * D_ * 2);
  bf16* w1_t   = (bf16*)alloc((size_t)F_ * D_ * 2);
  bf16* w2_t   = (bf16*)alloc((size_t)D_ * F_ * 2);
  float* bqkv  = (float*)alloc((size_t)QKVN * 4);
  bf16* x1   = (bf16*)alloc((size_t)BT * D_ * 2);
  bf16* qkv  = (bf16*)alloc((size_t)BT * QKVN * 2);
  bf16* vTb  = (bf16*)alloc((size_t)BT * D_ * 2);
  bf16* ctx  = (bf16*)alloc((size_t)BT * D_ * 2);
  float* xr  = (float*)alloc((size_t)BT * D_ * 4);
  bf16* yb   = (bf16*)alloc((size_t)BT * D_ * 2);
  bf16* hbuf = (bf16*)alloc((size_t)BT * F_ * 2);

  const int g_dd = (D_ * D_ + 255) / 256;
  const int g_df = (D_ * F_ + 255) / 256;
  tconv_kernel<<<g_dd, 256, 0, stream>>>(wq, wqkv_t,                 D_, D_);
  tconv_kernel<<<g_dd, 256, 0, stream>>>(wk, wqkv_t + (size_t)D_ * D_,     D_, D_);
  tconv_kernel<<<g_dd, 256, 0, stream>>>(wv, wqkv_t + (size_t)2 * D_ * D_, D_, D_);
  tconv_kernel<<<g_dd, 256, 0, stream>>>(wo, wo_t, D_, D_);
  tconv_kernel<<<g_df, 256, 0, stream>>>(w1, w1_t, D_, F_);
  tconv_kernel<<<g_df, 256, 0, stream>>>(w2, w2_t, F_, D_);
  bcat_kernel<<<6, 256, 0, stream>>>(bq, bk, bv, bqkv);

  // LN1 -> x1 (bf16)
  ln_kernel<<<BT, 256, 0, stream>>>(inputs, ln1_s, ln1_o, x1);
  // QKV projection: [8192,512] @ [512,1536]
  gemm_bf16_kernel<<<dim3(QKVN / 128, BT / 128), 256, 0, stream>>>(
      x1, wqkv_t, bqkv, nullptr, nullptr, qkv, BT, QKVN, D_, 0);
  // transpose V per head for PV B-fragments
  vtrans_kernel<<<(BT * D_) / 256, 256, 0, stream>>>(qkv, vTb);
  // flash attention: 16 (b,h) slices x 32 q-blocks
  attn_kernel<<<B_ * H_ * (T_ / 128), 256, 0, stream>>>(qkv, vTb, ctx);
  // output projection + residual -> x (fp32)
  gemm_bf16_kernel<<<dim3(D_ / 128, BT / 128), 256, 0, stream>>>(
      ctx, wo_t, bo, inputs, xr, nullptr, BT, D_, D_, 0);
  // LN2 -> y (bf16)
  ln_kernel<<<BT, 256, 0, stream>>>(xr, ln2_s, ln2_o, yb);
  // FFN1 with GELU -> h (bf16)
  gemm_bf16_kernel<<<dim3(F_ / 128, BT / 128), 256, 0, stream>>>(
      yb, w1_t, b1, nullptr, nullptr, hbuf, BT, F_, D_, 1);
  // FFN2 + residual -> out (fp32)
  gemm_bf16_kernel<<<dim3(D_ / 128, BT / 128), 256, 0, stream>>>(
      hbuf, w2_t, b2, xr, out, nullptr, BT, D_, F_, 0);
}
